// Defog_78606491451947
// MI455X (gfx1250) — matrix operand order repaired and checked by reading, not compiled, not run on hardware
//
#include <hip/hip_runtime.h>
#include <hip/hip_bf16.h>
#include <math.h>

typedef __attribute__((ext_vector_type(16))) _Float16 v16h;
typedef __attribute__((ext_vector_type(8)))  float    v8f;

// ---------------- problem constants ----------------
#define NIMG   16
#define ENC    256
#define MID    128      // ENC/2
#define HW     512
#define NPIX   (HW*HW)          // 262144 per image
#define KTOT   (ENC*9)          // 2304
#define NCHUNK (KTOT/32)        // 72 K-chunks of 32
#define TOPK   262              // max(int(262144*0.001),1)
#define NBINS  1024

// ---------------- workspace layout (in floats) ----------------
// h1   : 16*128*16*16 = 524288
// dc2/T: 16*512*512   = 4194304   (aliased by packed-A during conv stage)
// tmp  : 16*512*512   = 4194304   (first 524288 floats also aliased by packed-A)
// small: params/thresh/sumA/A/cnt/minmax/hist
// Bh   : 128*2304 halfs = 147456 floats
#define F_H1     0
#define F_DC2    (524288)
#define F_TMP    (F_DC2 + 4194304)
#define F_SMALL  (F_TMP + 4194304)
#define F_PARAMS (F_SMALL + 0)     // 16 f
#define F_THRESH (F_SMALL + 16)    // 16 f
#define F_SUMA   (F_SMALL + 32)    // 48 f
#define F_A      (F_SMALL + 80)    // 48 f
#define F_CNT    (F_SMALL + 128)   // 16 u32
#define F_MINMAX (F_SMALL + 144)   // 2 u32
#define F_HIST   (F_SMALL + 160)   // 16*1024 u32
#define F_BH     (F_SMALL + 16544) // 147456 floats (294912 halfs)
// packed A (im2col, fragment layout): 16*16*72*32*16 halfs = 9437184 halfs
// = 4718592 float-slots, aliased onto [F_DC2, F_DC2+4718592): dc2 + head of tmp.
// Safe: packed-A is dead before dc2/tmp are first written (single-stream order).
#define F_AH     F_DC2

__device__ __forceinline__ unsigned f2ord(float f) {
    unsigned u = __float_as_uint(f);
    return (u & 0x80000000u) ? ~u : (u | 0x80000000u);
}
__device__ __forceinline__ float ord2f(unsigned u) {
    return __uint_as_float((u & 0x80000000u) ? (u & 0x7FFFFFFFu) : ~u);
}

// ---------------- 0: reset accumulators (graph-safe determinism) ----------------
__global__ void init_kernel(unsigned* hist, float* sumA, unsigned* cntA, unsigned* minmax) {
    int i = blockIdx.x * blockDim.x + threadIdx.x;
    for (int j = i; j < NIMG * NBINS; j += gridDim.x * blockDim.x) hist[j] = 0u;
    if (i < 48) sumA[i] = 0.0f;
    if (i < 16) cntA[i] = 0u;
    if (i == 0) { minmax[0] = 0xFFFFFFFFu; minmax[1] = 0u; }
}

// ---------------- P1: pack W1 -> f16 [oc][k] ----------------
__global__ void pack_W_kernel(const float* __restrict__ W1, _Float16* __restrict__ Bh) {
    int i = (blockIdx.x * blockDim.x + threadIdx.x) * 4;   // 4 per thread
    if (i >= MID * KTOT) return;
    #pragma unroll
    for (int j = 0; j < 4; ++j) Bh[i + j] = (_Float16)W1[i + j];
}

// ---------------- P2: im2col -> f16 in WMMA lane-fragment layout (contiguous k) ----------------
// Ah[n][mt][chunk][lane][e], e->k mapping (contiguous k order, matched by the B reads):
//   row = lane&15, k = k0 + ((lane>>4)<<4) + e
__global__ __launch_bounds__(256) void pack_A_kernel(const float* __restrict__ latent,
                                                     _Float16* __restrict__ Ah) {
    int flat = blockIdx.x * blockDim.x + threadIdx.x;  // 0 .. 16*16*72*32-1
    int lane  = flat & 31;
    int rest  = flat >> 5;
    int chunk = rest % NCHUNK;
    rest /= NCHUNK;
    int mt = rest & 15;
    int n  = rest >> 4;

    int row = lane & 15;
    int p  = mt * 16 + row;
    int oy = p >> 4, ox = p & 15;
    int k0 = chunk * 32 + ((lane >> 4) << 4);

    v16h frag;
    #pragma unroll
    for (int e = 0; e < 16; ++e) {
        int k = k0 + e;
        int ic = k / 9, rem = k - ic * 9;
        int ky = rem / 3, kx = rem - ky * 3;
        int iy = 2 * oy - 1 + ky, ix = 2 * ox - 1 + kx;
        float v = 0.0f;
        if ((unsigned)iy < 32u && (unsigned)ix < 32u)
            v = latent[((n * ENC + ic) * 32 + iy) * 32 + ix];
        frag[e] = (_Float16)v;
    }
    *(v16h*)&Ah[(size_t)flat * 16] = frag;   // 32B aligned contiguous store
}

// ---------------- 1: conv1 GEMM — pure load/wmma hot loop ----------------
// grid: 16 images * 16 M-tiles; block: 256 thr = 8 waves (one 16-chan N-tile each)
__global__ __launch_bounds__(256) void conv1_wmma_kernel(
    const _Float16* __restrict__ Ah, const _Float16* __restrict__ Bh,
    const float* __restrict__ b1, float* __restrict__ h1)
{
    const int n    = blockIdx.x >> 4;
    const int mt   = blockIdx.x & 15;
    const int wv   = threadIdx.x >> 5;
    const int lane = threadIdx.x & 31;
    const int oc   = wv * 16 + (lane & 15);

    const _Float16* aPtr = Ah + (((size_t)(n * 16 + mt) * NCHUNK) * 32 + lane) * 16;
    const _Float16* bPtr = Bh + (size_t)oc * KTOT + ((lane >> 4) << 4);

    v8f acc = {};
    #pragma unroll 8
    for (int c = 0; c < NCHUNK; ++c) {
        __builtin_prefetch((const void*)(aPtr + 2 * 512), 0, 1);  // global_prefetch_b8
        v16h a = *(const v16h*)aPtr;    // 32B aligned A fragment
        v16h b = *(const v16h*)bPtr;    // 32B aligned B fragment
        acc = __builtin_amdgcn_wmma_f32_16x16x32_f16(
            false, a, false, b, (short)0, acc, false, false);
        aPtr += 512;                    // next chunk: 32 lanes * 16 halfs
        bPtr += 32;
    }

    // D layout: VGPR i -> row = i + (lane>=16 ? 8:0), col = lane&15. Bias + LeakyReLU.
    const int m0 = mt * 16;
    float bias = b1[oc];
    #pragma unroll
    for (int i = 0; i < 8; ++i) {
        int row = i + ((lane >> 4) << 3);
        int p = m0 + row;
        int oy = p >> 4, ox = p & 15;
        float v = acc[i] + bias;
        v = (v >= 0.0f) ? v : 0.02f * v;
        h1[((n * MID + oc) * 16 + oy) * 16 + ox] = v;
    }
}

// ---------------- 2: conv2 + mean + 1x1 conv + tanh ----------------
__global__ void head_kernel(const float* __restrict__ h1,
                            const float* __restrict__ W2, const float* __restrict__ b2,
                            const float* __restrict__ W3, const float* __restrict__ b3,
                            float* __restrict__ params)
{
    int n = blockIdx.x;
    int t = threadIdx.x;               // 64 threads -> 8x8 output
    int oy = t >> 3, ox = t & 7;
    float s = b2[0];
    for (int ic = 0; ic < MID; ++ic) {
        const float* hb = h1 + ((size_t)(n * MID + ic)) * 256;
        #pragma unroll
        for (int ky = 0; ky < 3; ++ky) {
            int iy = 2 * oy - 1 + ky;
            if ((unsigned)iy >= 16u) continue;
            #pragma unroll
            for (int kx = 0; kx < 3; ++kx) {
                int ix = 2 * ox - 1 + kx;
                if ((unsigned)ix >= 16u) continue;
                s += hb[iy * 16 + ix] * W2[(ic * 3 + ky) * 3 + kx];
            }
        }
    }
    __shared__ float red[64];
    red[t] = s;
    __syncthreads();
    for (int off = 32; off > 0; off >>= 1) {
        if (t < off) red[t] += red[t + off];
        __syncthreads();
    }
    if (t == 0) {
        float m = red[0] * (1.0f / 64.0f);
        float v = m * W3[0] + b3[0];
        params[n] = tanhf(v) * 0.5f + 0.5f;   // VALUE_RANGE (0,1)
    }
}

// ---------------- 3: dark channel histogram ----------------
__global__ void dark_hist_kernel(const float* __restrict__ x, unsigned* __restrict__ hist)
{
    __shared__ unsigned sh[NBINS];
    for (int i = threadIdx.x; i < NBINS; i += blockDim.x) sh[i] = 0u;
    __syncthreads();
    size_t gid = (size_t)blockIdx.x * blockDim.x + threadIdx.x;
    int n = (int)(gid >> 18);
    int p = (int)(gid & (NPIX - 1));
    const float* xb = x + (size_t)n * 3 * NPIX;
    float d = fminf(xb[p], fminf(xb[NPIX + p], xb[2 * NPIX + p]));
    int bin = (int)(d * (float)NBINS);
    bin = bin < 0 ? 0 : (bin > NBINS - 1 ? NBINS - 1 : bin);
    atomicAdd(&sh[bin], 1u);
    __syncthreads();
    unsigned* gh = hist + n * NBINS;   // whole block is in one image (NPIX % 256 == 0)
    for (int i = threadIdx.x; i < NBINS; i += blockDim.x)
        if (sh[i]) atomicAdd(&gh[i], sh[i]);
}

// ---------------- 4: threshold = top-0.1% cutoff bin ----------------
__global__ void thresh_kernel(const unsigned* __restrict__ hist, float* __restrict__ thresh)
{
    int n = threadIdx.x;
    if (n >= NIMG) return;
    const unsigned* h = hist + n * NBINS;
    unsigned acc = 0;
    int b = NBINS - 1;
    for (; b > 0; --b) { acc += h[b]; if (acc >= TOPK) break; }
    thresh[n] = (float)b / (float)NBINS;
}

// ---------------- 5: accumulate atmosphere A over bright-dark pixels ----------------
__global__ void accum_A_kernel(const float* __restrict__ x, const float* __restrict__ thresh,
                               float* __restrict__ sumA, unsigned* __restrict__ cntA)
{
    __shared__ float s0, s1, s2;
    __shared__ unsigned sc;
    if (threadIdx.x == 0) { s0 = s1 = s2 = 0.0f; sc = 0u; }
    __syncthreads();
    size_t gid = (size_t)blockIdx.x * blockDim.x + threadIdx.x;
    int n = (int)(gid >> 18);
    int p = (int)(gid & (NPIX - 1));
    const float* xb = x + (size_t)n * 3 * NPIX;
    float r = xb[p], g = xb[NPIX + p], bl = xb[2 * NPIX + p];
    float d = fminf(r, fminf(g, bl));
    if (d >= thresh[n]) {
        atomicAdd(&s0, r); atomicAdd(&s1, g); atomicAdd(&s2, bl); atomicAdd(&sc, 1u);
    }
    __syncthreads();
    if (threadIdx.x == 0 && sc > 0u) {
        atomicAdd(&sumA[n * 3 + 0], s0);
        atomicAdd(&sumA[n * 3 + 1], s1);
        atomicAdd(&sumA[n * 3 + 2], s2);
        atomicAdd(&cntA[n], sc);
    }
}

__global__ void finalize_A_kernel(const float* __restrict__ sumA, const unsigned* __restrict__ cntA,
                                  float* __restrict__ A)
{
    int i = threadIdx.x;
    if (i < 48) {
        unsigned c = cntA[i / 3];
        A[i] = sumA[i] / (float)(c ? c : 1u);
    }
}

// ---------------- 6: dc2 = min_c(x / A) ----------------
__global__ void dc2_kernel(const float* __restrict__ x, const float* __restrict__ A,
                           float* __restrict__ dc2)
{
    size_t gid = (size_t)blockIdx.x * blockDim.x + threadIdx.x;
    int n = (int)(gid >> 18);
    int p = (int)(gid & (NPIX - 1));
    const float* xb = x + (size_t)n * 3 * NPIX;
    float a0 = A[n * 3 + 0], a1 = A[n * 3 + 1], a2 = A[n * 3 + 2];
    float v = fminf(xb[p] / a0, fminf(xb[NPIX + p] / a1, xb[2 * NPIX + p] / a2));
    dc2[(size_t)n * NPIX + p] = v;
}

// ---------------- 7: separable 7x7 min-pool, pass 1 (rows) ----------------
__global__ void rowmin_kernel(const float* __restrict__ dc2, float* __restrict__ tmp)
{
    size_t gid = (size_t)blockIdx.x * blockDim.x + threadIdx.x;
    int n = (int)(gid >> 18);
    int p = (int)(gid & (NPIX - 1));
    int y = p >> 9, xc = p & (HW - 1);
    const float* base = dc2 + (size_t)n * NPIX + (size_t)y * HW;
    float m = INFINITY;
    #pragma unroll
    for (int dx = -3; dx <= 3; ++dx) {
        int xx = xc + dx;
        if ((unsigned)xx < (unsigned)HW) m = fminf(m, base[xx]);
    }
    tmp[(size_t)n * NPIX + p] = m;
}

// ---------------- 8: pass 2 (cols) fused with T = max(1 - params*dc2, 0.01) ----------------
__global__ void colmin_T_kernel(const float* __restrict__ tmp, const float* __restrict__ params,
                                float* __restrict__ T)
{
    size_t gid = (size_t)blockIdx.x * blockDim.x + threadIdx.x;
    int n = (int)(gid >> 18);
    int p = (int)(gid & (NPIX - 1));
    int y = p >> 9, xc = p & (HW - 1);
    const float* base = tmp + (size_t)n * NPIX;
    float m = INFINITY;
    #pragma unroll
    for (int dy = -3; dy <= 3; ++dy) {
        int yy = y + dy;
        if ((unsigned)yy < (unsigned)HW) m = fminf(m, base[(size_t)yy * HW + xc]);
    }
    float t = 1.0f - params[n] * m;
    T[(size_t)n * NPIX + p] = fmaxf(t, 0.01f);
}

// ---------------- 9: reconstruct + global min/max ----------------
__global__ void recon_kernel(const float* __restrict__ x, const float* __restrict__ A,
                             const float* __restrict__ T, float* __restrict__ out,
                             unsigned* __restrict__ minmax)
{
    size_t gid = (size_t)blockIdx.x * blockDim.x + threadIdx.x;
    int n = (int)(gid >> 18);
    int p = (int)(gid & (NPIX - 1));
    float Tv = T[(size_t)n * NPIX + p];
    float inv = 1.0f / Tv;
    float lmn = INFINITY, lmx = -INFINITY;
    #pragma unroll
    for (int c = 0; c < 3; ++c) {
        float a = A[n * 3 + c];
        size_t idx = ((size_t)n * 3 + c) * NPIX + p;
        float o = (x[idx] - a) * inv + a;
        out[idx] = o;
        lmn = fminf(lmn, o);
        lmx = fmaxf(lmx, o);
    }
    __shared__ float smn[256], smx[256];
    int t = threadIdx.x;
    smn[t] = lmn; smx[t] = lmx;
    __syncthreads();
    for (int off = 128; off > 0; off >>= 1) {
        if (t < off) {
            smn[t] = fminf(smn[t], smn[t + off]);
            smx[t] = fmaxf(smx[t], smx[t + off]);
        }
        __syncthreads();
    }
    if (t == 0) {
        atomicMin(&minmax[0], f2ord(smn[0]));
        atomicMax(&minmax[1], f2ord(smx[0]));
    }
}

// ---------------- 10: global min-max normalize ----------------
__global__ void normalize_kernel(float* __restrict__ out, const unsigned* __restrict__ minmax,
                                 size_t total)
{
    float mn = ord2f(minmax[0]);
    float mx = ord2f(minmax[1]);
    float inv = 1.0f / (mx - mn);
    size_t stride = (size_t)gridDim.x * blockDim.x;
    for (size_t i = (size_t)blockIdx.x * blockDim.x + threadIdx.x; i < total; i += stride)
        out[i] = (out[i] - mn) * inv;
}

// ---------------- host-side launcher ----------------
extern "C" void kernel_launch(void* const* d_in, const int* in_sizes, int n_in,
                              void* d_out, int out_size, void* d_ws, size_t ws_size,
                              hipStream_t stream) {
    const float* x      = (const float*)d_in[0];
    const float* latent = (const float*)d_in[1];
    const float* W1     = (const float*)d_in[2];
    const float* b1     = (const float*)d_in[3];
    const float* W2     = (const float*)d_in[4];
    const float* b2     = (const float*)d_in[5];
    const float* W3     = (const float*)d_in[6];
    const float* b3     = (const float*)d_in[7];
    float* out = (float*)d_out;

    float*     ws     = (float*)d_ws;
    float*     h1     = ws + F_H1;
    float*     dc2    = ws + F_DC2;     // later reused as T
    float*     tmp    = ws + F_TMP;
    float*     params = ws + F_PARAMS;
    float*     thresh = ws + F_THRESH;
    float*     sumA   = ws + F_SUMA;
    float*     A      = ws + F_A;
    unsigned*  cntA   = (unsigned*)(ws + F_CNT);
    unsigned*  minmax = (unsigned*)(ws + F_MINMAX);
    unsigned*  hist   = (unsigned*)(ws + F_HIST);
    _Float16*  Bh     = (_Float16*)(ws + F_BH);
    _Float16*  Ah     = (_Float16*)(ws + F_AH);   // aliases dc2+tmp head; dead before they're written

    const int pixBlocks = (NIMG * NPIX) / 256;    // 16384

    init_kernel<<<64, 256, 0, stream>>>(hist, sumA, cntA, minmax);

    // conv stage: pack then pure-WMMA GEMM
    pack_W_kernel<<<(MID * KTOT / 4 + 255) / 256, 256, 0, stream>>>(W1, Bh);
    pack_A_kernel<<<(NIMG * 16 * NCHUNK * 32) / 256, 256, 0, stream>>>(latent, Ah);
    conv1_wmma_kernel<<<NIMG * 16, 256, 0, stream>>>(Ah, Bh, b1, h1);
    head_kernel<<<NIMG, 64, 0, stream>>>(h1, W2, b2, W3, b3, params);

    // atmosphere estimation
    dark_hist_kernel<<<pixBlocks, 256, 0, stream>>>(x, hist);
    thresh_kernel<<<1, 32, 0, stream>>>(hist, thresh);
    accum_A_kernel<<<pixBlocks, 256, 0, stream>>>(x, thresh, sumA, cntA);
    finalize_A_kernel<<<1, 64, 0, stream>>>(sumA, cntA, A);

    // transmission map (packed-A is dead from here; dc2/tmp may be written)
    dc2_kernel<<<pixBlocks, 256, 0, stream>>>(x, A, dc2);
    rowmin_kernel<<<pixBlocks, 256, 0, stream>>>(dc2, tmp);
    colmin_T_kernel<<<pixBlocks, 256, 0, stream>>>(tmp, params, dc2 /* becomes T */);

    // reconstruction + global min-max normalize
    recon_kernel<<<pixBlocks, 256, 0, stream>>>(x, A, dc2, out, minmax);
    normalize_kernel<<<2048, 256, 0, stream>>>(out, minmax, (size_t)NIMG * 3 * NPIX);
}
